// RecurrentLSTMStack_23450521436741
// MI455X (gfx1250) — matrix operand order repaired and checked by reading, not compiled, not run on hardware
//
#include <hip/hip_runtime.h>
#include <hip/hip_bf16.h>

#define T_STEPS 256
#define BATCH   2048
#define DIN     16
#define HID     64
#define G4      (4 * HID)   // 256 gate rows
#define BT      32          // batch tile per workgroup -> 64 workgroups
#define NTHREADS 256        // 8 wave32s
#define XPT     ((BT * DIN) / NTHREADS)   // x elements per thread = 2

typedef __attribute__((ext_vector_type(16))) __bf16 v16bf;
typedef __attribute__((ext_vector_type(8)))  float  v8f;

union FragBF { uint4 q[2]; v16bf v; };

__device__ __forceinline__ unsigned short f2bf(float f) {
  unsigned int u = __float_as_uint(f);
  unsigned int r = u + 0x7FFFu + ((u >> 16) & 1u);   // round-to-nearest-even
  return (unsigned short)(r >> 16);
}
// fast rcp (v_rcp_f32) instead of IEEE div sequence: keeps the serial
// elementwise path in cheap TRANS ops that co-execute with WMMA.
__device__ __forceinline__ float fast_sigmoid(float x) {
  return __builtin_amdgcn_rcpf(1.0f + __expf(-x));
}
__device__ __forceinline__ float fast_tanh(float x) {
  float e = __expf(2.0f * x);
  return 1.0f - 2.0f * __builtin_amdgcn_rcpf(e + 1.0f);  // saturates at +/-1
}

// A fragment: 16x32 bf16 tile from LDS, row-major, leading dim ldk (elements).
// lanes 0-15 -> M=0..15, VGPR0..3 = K 0..7, VGPR4..7 = K 16..23;
// lanes 16-31 same M, K offset +8.  => two contiguous 16B LDS loads per lane.
__device__ __forceinline__ v16bf load_a(const unsigned short* base, int ldk,
                                        int row0, int k0, int lane) {
  int m = lane & 15, half = lane >> 4;
  const unsigned short* p = base + (row0 + m) * ldk + k0 + half * 8;
  FragBF f;
  f.q[0] = *(const uint4*)(p);        // K offsets 0..7   -> VGPR 0..3
  f.q[1] = *(const uint4*)(p + 16);   // K offsets 16..23 -> VGPR 4..7
  return f.v;
}
// B fragment: 32x16 bf16, B[k][n] = W[col0+n][k0+k], W row-major.
// lanes 0-15 hold K=0..7,16..23 for column n, lanes 16-31 hold K=8..15,24..31.
__device__ __forceinline__ v16bf load_b(const unsigned short* base, int ldk,
                                        int col0, int k0, int lane) {
  int n = lane & 15, khalf = lane >> 4;
  const unsigned short* p = base + (col0 + n) * ldk + k0 + khalf * 8;
  FragBF f;
  f.q[0] = *(const uint4*)(p);        // K offsets 0..7   -> VGPR 0..3
  f.q[1] = *(const uint4*)(p + 16);   // K offsets 16..23 -> VGPR 4..7
  return f.v;
}

__global__ __launch_bounds__(NTHREADS)
void lstm3_wmma_kernel(const float* __restrict__ x,
                       const float* __restrict__ Wih0, const float* __restrict__ Whh0,
                       const float* __restrict__ bih0, const float* __restrict__ bhh0,
                       const float* __restrict__ Wih1, const float* __restrict__ Whh1,
                       const float* __restrict__ bih1, const float* __restrict__ bhh1,
                       const float* __restrict__ Wih2, const float* __restrict__ Whh2,
                       const float* __restrict__ bih2, const float* __restrict__ bhh2,
                       float* __restrict__ out)
{
  // ---- LDS: weights (bf16), fused biases (f32), h state (bf16), x staging ----
  __shared__ __align__(16) unsigned short s_wih0[G4 * 32];     // K padded 16->32
  __shared__ __align__(16) unsigned short s_whh0[G4 * HID];
  __shared__ __align__(16) unsigned short s_wih1[G4 * HID];
  __shared__ __align__(16) unsigned short s_whh1[G4 * HID];
  __shared__ __align__(16) unsigned short s_wih2[G4 * HID];
  __shared__ __align__(16) unsigned short s_whh2[G4 * HID];
  __shared__ __align__(16) float          s_bias[3][G4];
  __shared__ __align__(16) unsigned short s_h[3][BT * HID];
  __shared__ __align__(16) unsigned short s_x[BT * 32];        // K padded 16->32

  const int tid  = threadIdx.x;
  const int lane = tid & 31;
  const int wave = tid >> 5;
  const int wgb  = blockIdx.x * BT;

  // ---- one-time staging: fp32 weights -> bf16 LDS ----
  for (int i = tid; i < G4 * 32; i += NTHREADS) {
    int r = i >> 5, c = i & 31;
    s_wih0[i] = (c < DIN) ? f2bf(Wih0[r * DIN + c]) : (unsigned short)0;
  }
  for (int i = tid; i < G4 * HID; i += NTHREADS) {
    s_whh0[i] = f2bf(Whh0[i]);
    s_wih1[i] = f2bf(Wih1[i]);
    s_whh1[i] = f2bf(Whh1[i]);
    s_wih2[i] = f2bf(Wih2[i]);
    s_whh2[i] = f2bf(Whh2[i]);
  }
  for (int i = tid; i < G4; i += NTHREADS) {
    s_bias[0][i] = bih0[i] + bhh0[i];
    s_bias[1][i] = bih1[i] + bhh1[i];
    s_bias[2][i] = bih2[i] + bhh2[i];
  }
  for (int i = tid; i < BT * HID; i += NTHREADS) {
    s_h[0][i] = 0; s_h[1][i] = 0; s_h[2][i] = 0;
  }
  for (int i = tid; i < BT * 32; i += NTHREADS) s_x[i] = 0;

  // c-state in registers: this wave's fixed C-fragment slots, per layer
  float cst[3][8];
#pragma unroll
  for (int l = 0; l < 3; ++l)
#pragma unroll
    for (int r = 0; r < 8; ++r) cst[l][r] = 0.0f;

  // wave tiling of the (32 x 256) gate matrix: 2 row-tiles x 16 col-tiles.
  // Unit = (row-tile, j-col-group) = 4 gate tiles (i,f,g,o share lane slots).
  // 8 units -> exactly 1 per wave.
  const int cgrp = wave & 3;        // hidden cols [16*cgrp, 16*cgrp+16)
  const int rt   = wave >> 2;       // row tile 0/1
  const int arow = rt * 16;
  const int n    = lane & 15;
  const int half = lane >> 4;
  const int jcol = cgrp * 16 + n;

  // preload x_0 into registers (XPT elements per thread)
  float xv[XPT];
#pragma unroll
  for (int e = 0; e < XPT; ++e) {
    int i = tid + e * NTHREADS;
    int row = i >> 4, col = i & 15;
    xv[e] = x[((size_t)0 * BATCH + wgb + row) * DIN + col];
  }

  __syncthreads();

  for (int t = 0; t < T_STEPS; ++t) {
    // commit prefetched x_t registers -> bf16 LDS (pad cols stay zero)
#pragma unroll
    for (int e = 0; e < XPT; ++e) {
      int i = tid + e * NTHREADS;
      int row = i >> 4, col = i & 15;
      s_x[row * 32 + col] = f2bf(xv[e]);
    }
    __syncthreads();

    // issue x_{t+1} loads now; latency hidden behind 3 layers of compute
    if (t + 1 < T_STEPS) {
#pragma unroll
      for (int e = 0; e < XPT; ++e) {
        int i = tid + e * NTHREADS;
        int row = i >> 4, col = i & 15;
        xv[e] = x[((size_t)(t + 1) * BATCH + wgb + row) * DIN + col];
      }
    }

#pragma unroll
    for (int l = 0; l < 3; ++l) {
      const unsigned short* s_in  = (l == 0) ? s_x : s_h[(l > 0) ? (l - 1) : 0];
      const int in_ldk            = (l == 0) ? 32 : HID;
      const int in_K              = (l == 0) ? 32 : HID;
      const unsigned short* w_in  = (l == 0) ? s_wih0 : ((l == 1) ? s_wih1 : s_wih2);
      const unsigned short* w_rec = (l == 0) ? s_whh0 : ((l == 1) ? s_whh1 : s_whh2);
      const unsigned short* s_rec = s_h[l];

      v8f acc[4];
#pragma unroll
      for (int g = 0; g < 4; ++g) {
        const int col = g * HID + cgrp * 16;   // gate g, this wave's hidden cols
        v8f a = {};
        for (int k0 = 0; k0 < in_K; k0 += 32) {            // input term
          v16bf af = load_a(s_in, in_ldk, arow, k0, lane);
          v16bf bf = load_b(w_in, in_ldk, col, k0, lane);
          a = __builtin_amdgcn_wmma_f32_16x16x32_bf16(
                  false, af, false, bf, (short)0, a, false, false);
        }
        for (int k0 = 0; k0 < HID; k0 += 32) {             // recurrent term
          v16bf af = load_a(s_rec, HID, arow, k0, lane);
          v16bf bf = load_b(w_rec, HID, col, k0, lane);
          a = __builtin_amdgcn_wmma_f32_16x16x32_bf16(
                  false, af, false, bf, (short)0, a, false, false);
        }
        acc[g] = a;
      }
      __syncthreads();   // all waves done READING s_h[l] / s_in before overwrite

      const float bi  = s_bias[l][0 * HID + jcol];
      const float bf_ = s_bias[l][1 * HID + jcol];
      const float bg  = s_bias[l][2 * HID + jcol];
      const float bo  = s_bias[l][3 * HID + jcol];

#pragma unroll
      for (int r = 0; r < 8; ++r) {
        float iv = fast_sigmoid(acc[0][r] + bi);
        float fv = fast_sigmoid(acc[1][r] + bf_);
        float gv = fast_tanh   (acc[2][r] + bg);
        float ov = fast_sigmoid(acc[3][r] + bo);
        float cn = fv * cst[l][r] + iv * gv;
        cst[l][r] = cn;
        float hn = ov * fast_tanh(cn);
        int m = arow + r + 8 * half;          // C-fragment row mapping
        s_h[l][m * HID + jcol] = f2bf(hn);
        if (l == 2) {
          out[((size_t)t * BATCH + wgb + m) * HID + jcol] = hn;
        }
      }
      __syncthreads();   // h_new visible before next layer / next step reads
    }
  }
}

extern "C" void kernel_launch(void* const* d_in, const int* in_sizes, int n_in,
                              void* d_out, int out_size, void* d_ws, size_t ws_size,
                              hipStream_t stream) {
  (void)in_sizes; (void)n_in; (void)d_ws; (void)ws_size; (void)out_size;
  const float* x    = (const float*)d_in[0];
  const float* Wih0 = (const float*)d_in[1];
  const float* Whh0 = (const float*)d_in[2];
  const float* bih0 = (const float*)d_in[3];
  const float* bhh0 = (const float*)d_in[4];
  const float* Wih1 = (const float*)d_in[5];
  const float* Whh1 = (const float*)d_in[6];
  const float* bih1 = (const float*)d_in[7];
  const float* bhh1 = (const float*)d_in[8];
  const float* Wih2 = (const float*)d_in[9];
  const float* Whh2 = (const float*)d_in[10];
  const float* bih2 = (const float*)d_in[11];
  const float* bhh2 = (const float*)d_in[12];
  float* out = (float*)d_out;

  dim3 grid(BATCH / BT);      // 64 workgroups, each owns a 32-row batch slice
  dim3 block(NTHREADS);       // 8 wave32s
  lstm3_wmma_kernel<<<grid, block, 0, stream>>>(
      x, Wih0, Whh0, bih0, bhh0,
      Wih1, Whh1, bih1, bhh1,
      Wih2, Whh2, bih2, bhh2, out);
}
